// EncoderLayer_90967407329637
// MI455X (gfx1250) — matrix operand order repaired and checked by reading, not compiled, not run on hardware
//
#include <hip/hip_runtime.h>

// ---------------------------------------------------------------------------
// CDNA5 (gfx1250) transformer encoder layer: LN -> MHA(causal) -> FFN
// B=32, L=512, D=512, H=8, head_dim=64.
// Matmuls via v_wmma_f32_16x16x32_f16; weight tiles staged to LDS with
// global_load_async_to_lds_b128 (ASYNCcnt), double-buffered.
// ---------------------------------------------------------------------------

typedef __attribute__((ext_vector_type(16))) _Float16 v16h;
typedef __attribute__((ext_vector_type(8)))  float    v8f;

static constexpr int   kTokens = 32 * 512;   // B*L
static constexpr int   kD      = 512;
static constexpr int   kSeq    = 512;
static constexpr int   kHeads  = 8;
static constexpr int   kHd     = 64;
static constexpr float kNeg    = -4294967295.0f;   // -(2^32 - 1)

// ---- CDNA5 async global->LDS helpers (inline asm: portable across toolchains)
static __device__ __forceinline__ unsigned lds_addr_of(const void* p) {
    // generic shared pointer: low 32 bits are the LDS byte address
    return (unsigned)(unsigned long long)p;
}
static __device__ __forceinline__ void async_copy_b128(void* lds, const void* gaddr) {
    asm volatile("global_load_async_to_lds_b128 %0, %1, off"
                 :: "v"(lds_addr_of(lds)), "v"(gaddr) : "memory");
}

// ---- WMMA fragment loaders (wave32; k = 16*(lane/16) + element in A and B) ----------

// A matrix 16x32 f16 (MxK): lanes 0-15 M=lane (K 0-15),
// lanes 16-31 M=lane-16 (K 16-31). VGPR v holds a K pair.
static __device__ __forceinline__ v16h
load_a_frag(const _Float16* A, int lda, int row0, int k0) {
    const int lane = threadIdx.x & 31;
    const _Float16* p = A + (size_t)(row0 + (lane & 15)) * lda + k0 + ((lane & 16) ? 16 : 0);
    v16h a;
#pragma unroll
    for (int v = 0; v < 4; ++v) { a[2 * v] = p[2 * v]; a[2 * v + 1] = p[2 * v + 1]; }
#pragma unroll
    for (int v = 0; v < 4; ++v) { a[8 + 2 * v] = p[8 + 2 * v]; a[9 + 2 * v] = p[8 + 2 * v + 1]; }
    return a;
}

// B matrix 32x16 f16 (KxN) with K contiguous at p (one row of W / K-tile):
// lanes 0-15: N=lane, K=0..15; lanes 16-31: N=lane-16, K=16..31.
static __device__ __forceinline__ v16h
load_b_frag_kcontig(const _Float16* W, int ldw, int n0, int k0) {
    const int lane = threadIdx.x & 31;
    const _Float16* p = W + (size_t)(n0 + (lane & 15)) * ldw + k0 + ((lane & 16) ? 16 : 0);
    v16h b;
#pragma unroll
    for (int e = 0; e < 16; ++e) b[e] = p[e];
    return b;
}

// B matrix 32x16 f16 where B[k][n] = V[k][n] row-major (n contiguous, k strided)
static __device__ __forceinline__ v16h
load_b_frag_ncontig(const _Float16* V, int ldv, size_t k0, int n0) {
    const int lane = threadIdx.x & 31;
    const _Float16* p = V + (k0 + ((lane & 16) ? 16 : 0)) * (size_t)ldv + n0 + (lane & 15);
    v16h b;
#pragma unroll
    for (int e = 0; e < 16; ++e) b[e] = p[(size_t)e * ldv];
    return b;
}

static __device__ __forceinline__ v8f
wmma_f16(v16h a, v16h b, v8f c) {
    return __builtin_amdgcn_wmma_f32_16x16x32_f16(false, a, false, b, (short)0, c, false, false);
}

// ---------------------------------------------------------------------------
// Kernel 1: LayerNorm + masks + f16 staging.  One wave per token row.
// ---------------------------------------------------------------------------
__global__ __launch_bounds__(128) void prep_kernel(
    const float* __restrict__ x, const float* __restrict__ gamma,
    const float* __restrict__ beta,
    float* __restrict__ xnf, _Float16* __restrict__ xnh, _Float16* __restrict__ xh,
    float* __restrict__ keymask, float* __restrict__ qmask) {
    const int lane  = threadIdx.x & 31;
    const int wave  = threadIdx.x >> 5;
    const int token = blockIdx.x * 4 + wave;
    if (token >= kTokens) return;
    const int l = token & (kSeq - 1);
    const float* xr = x + (size_t)token * kD;

    float s = 0.f, s2 = 0.f;
#pragma unroll
    for (int i = lane; i < kD; i += 32) { float v = xr[i]; s += v; s2 += v * v; }
#pragma unroll
    for (int o = 16; o; o >>= 1) { s += __shfl_xor(s, o); s2 += __shfl_xor(s2, o); }
    const float mean = s * (1.0f / kD);
    const float var  = fmaxf(s2 * (1.0f / kD) - mean * mean, 0.0f);
    const float rinv = rsqrtf(var + 1e-8f);   // eps inside sqrt, like reference

    float qs = 0.f;
#pragma unroll
    for (int i = lane; i < kD; i += 32) {
        const float v  = xr[i];
        const float xn = gamma[(size_t)l * kD + i] * ((v - mean) * rinv) + beta[(size_t)l * kD + i];
        xnf[(size_t)token * kD + i] = xn;
        xnh[(size_t)token * kD + i] = (_Float16)xn;
        xh[(size_t)token * kD + i]  = (_Float16)v;
        qs += xn;
    }
#pragma unroll
    for (int o = 16; o; o >>= 1) qs += __shfl_xor(qs, o);
    if (lane == 0) {
        keymask[token] = (s != 0.f) ? 1.f : 0.f;   // sign(|sum keys|)
        qmask[token]   = (qs != 0.f) ? 1.f : 0.f;  // sign(|sum queries|)
    }
}

// ---------------------------------------------------------------------------
// Kernel 2: f32 -> f16 conversion (weights)
// ---------------------------------------------------------------------------
__global__ __launch_bounds__(256) void cvt_kernel(const float* __restrict__ s,
                                                  _Float16* __restrict__ d, int n) {
    int i = blockIdx.x * blockDim.x + threadIdx.x;
    if (i < n) d[i] = (_Float16)s[i];
}

// ---------------------------------------------------------------------------
// Kernel 3: WMMA GEMM  C[M,N] = act(A[M,K] @ W[N,K]^T + bias) (+res) (*rowscale)
// block = 4 waves, 64x64 tile; wave = 16x64, K-step 32.
// Weight tile (64 rows x 32 halfs = 4KB) staged to LDS with async b128 copies,
// double-buffered: tile k+1 in flight while tile k feeds the WMMAs.
// mode 0: f16 out; 1: relu + f16 out; 2: +residual, *rowscale, f32 out
// ---------------------------------------------------------------------------
__global__ __launch_bounds__(128) void gemm_kernel(
    const _Float16* __restrict__ A, const _Float16* __restrict__ W,
    const float* __restrict__ bias,
    const float* __restrict__ residual, const float* __restrict__ rowscale,
    _Float16* __restrict__ outH, float* __restrict__ outF,
    int M, int N, int K, int mode) {
    __shared__ _Float16 bs[2][64 * 32];      // 2 x 4KB weight tiles

    const int lane = threadIdx.x & 31;
    const int wave = threadIdx.x >> 5;
    const int tid  = threadIdx.x;
    const int nt   = N >> 6;
    const int m0   = (blockIdx.x / nt) * 64 + wave * 16;
    const int n0   = (blockIdx.x % nt) * 64;

    // Stage one 64x32-half W tile into LDS: 256 16-byte chunks, 2 per thread.
    // chunk idx -> row = idx/4, 8-half column group = idx%4.
    auto issue_tile = [&](int k0, int buf) {
#pragma unroll
        for (int u = 0; u < 2; ++u) {
            const int idx = tid + u * 128;
            const int r = idx >> 2, c = idx & 3;
            async_copy_b128(&bs[buf][r * 32 + c * 8],
                            W + (size_t)(n0 + r) * K + k0 + c * 8);
        }
    };

    const int nk = K / 32;
    issue_tile(0, 0);

    v8f acc[4] = {v8f{}, v8f{}, v8f{}, v8f{}};
    for (int kt = 0; kt < nk; ++kt) {
        const int k0 = kt * 32;
        if (kt + 1 < nk) {
            issue_tile(k0 + 32, (kt + 1) & 1);                  // overlap next tile
            asm volatile("s_wait_asynccnt 2" ::: "memory");     // tile kt resident (mine)
        } else {
            asm volatile("s_wait_asynccnt 0" ::: "memory");
        }
        __syncthreads();                                        // tile kt resident (all waves)

        const v16h a = load_a_frag(A, K, m0, k0);
        const _Float16* bbuf = bs[kt & 1];
#pragma unroll
        for (int t = 0; t < 4; ++t) {
            const v16h b = load_b_frag_kcontig(bbuf, 32, 16 * t, 0);
            acc[t] = wmma_f16(a, b, acc[t]);
        }
        __syncthreads();                                        // readers done before reuse
    }

    const int coll  = lane & 15;
    const int rbase = m0 + ((lane & 16) ? 8 : 0);
#pragma unroll
    for (int t = 0; t < 4; ++t) {
        const int   col = n0 + 16 * t + coll;
        const float bv  = bias ? bias[col] : 0.f;
#pragma unroll
        for (int r = 0; r < 8; ++r) {
            const int row = rbase + r;
            float v = acc[t][r] + bv;
            if (mode == 1) v = v > 0.f ? v : 0.f;
            if (mode == 2) {
                v += residual[(size_t)row * N + col];
                v *= rowscale[row];
            }
            if (outH) outH[(size_t)row * N + col] = (_Float16)v;
            if (outF) outF[(size_t)row * N + col] = v;
        }
    }
}

// ---------------------------------------------------------------------------
// Kernel 4: attention. One wave per (batch, head, 16-query tile).
// scores = QK^T/8 (2 chained WMMAs, K=64) -> LDS -> masked softmax -> att@V.
// ---------------------------------------------------------------------------
__global__ __launch_bounds__(32) void attn_kernel(
    const _Float16* __restrict__ Qh, const _Float16* __restrict__ Kh,
    const _Float16* __restrict__ Vh,
    const float* __restrict__ keymask, const float* __restrict__ qmask,
    const float* __restrict__ xnf,
    float* __restrict__ attnF, _Float16* __restrict__ attnH) {
    __shared__ float    sc[16 * kSeq];       // 32 KB raw scores
    __shared__ _Float16 ah[16 * kSeq];       // 16 KB softmaxed probs (f16)

    const int lane = threadIdx.x & 31;
    const int qt   = blockIdx.x & 31;            // 32 query tiles
    const int h    = (blockIdx.x >> 5) & 7;      // 8 heads
    const int b    = blockIdx.x >> 8;            // 32 batches
    const int q0   = qt * 16;
    const int tok0 = b * kSeq;

    // Q fragments for the whole 64-wide head dim (two K=32 chunks)
    const v16h qa0 = load_a_frag(Qh, kD, tok0 + q0, h * kHd);
    const v16h qa1 = load_a_frag(Qh, kD, tok0 + q0, h * kHd + 32);

    const int rb = (lane & 16) ? 8 : 0;
    for (int kt = 0; kt < kSeq / 16; ++kt) {
        const int k0 = kt * 16;
        const v16h kb0 = load_b_frag_kcontig(Kh, kD, tok0 + k0, h * kHd);
        const v16h kb1 = load_b_frag_kcontig(Kh, kD, tok0 + k0, h * kHd + 32);
        v8f s = v8f{};
        s = wmma_f16(qa0, kb0, s);
        s = wmma_f16(qa1, kb1, s);
        const int   key = k0 + (lane & 15);
        const float km  = keymask[tok0 + key];
#pragma unroll
        for (int r = 0; r < 8; ++r) {
            const int q = q0 + rb + r;
            float v = s[r] * 0.125f;             // 1/sqrt(64)
            if (km == 0.f || key > q) v = kNeg;  // key-pad + causal masks
            sc[(rb + r) * kSeq + key] = v;
        }
    }

    // masked softmax: two lanes per row (same wave -> DS in-order, no barrier)
    {
        const int row  = lane >> 1;
        const int half = lane & 1;
        const float* rp = sc + row * kSeq;
        const int j0 = half * (kSeq / 2), j1 = j0 + kSeq / 2;
        float m = -3.4e38f;
        for (int j = j0; j < j1; ++j) m = fmaxf(m, rp[j]);
        m = fmaxf(m, __shfl_xor(m, 1));
        float ssum = 0.f;
        for (int j = j0; j < j1; ++j) ssum += __expf(rp[j] - m);
        ssum += __shfl_xor(ssum, 1);
        const float inv = 1.0f / ssum;
        const float qm  = qmask[tok0 + q0 + row];   // query mask AFTER softmax
        for (int j = j0; j < j1; ++j)
            ah[row * kSeq + j] = (_Float16)(__expf(rp[j] - m) * inv * qm);
    }

    // att @ V : A from LDS, B row-major V (n contiguous)
    v8f acc[4] = {v8f{}, v8f{}, v8f{}, v8f{}};
    for (int k0 = 0; k0 < kSeq; k0 += 32) {
        const v16h a = load_a_frag(ah, kSeq, 0, k0);
#pragma unroll
        for (int t = 0; t < 4; ++t) {
            const v16h bfr = load_b_frag_ncontig(Vh, kD, (size_t)tok0 + k0, h * kHd + 16 * t);
            acc[t] = wmma_f16(a, bfr, acc[t]);
        }
    }

    // epilogue: + x_norm residual, dual f32/f16 store
#pragma unroll
    for (int t = 0; t < 4; ++t) {
        const int col = h * kHd + 16 * t + (lane & 15);
#pragma unroll
        for (int r = 0; r < 8; ++r) {
            const size_t row = (size_t)tok0 + q0 + rb + r;
            const float  v   = acc[t][r] + xnf[row * kD + col];
            attnF[row * kD + col] = v;
            attnH[row * kD + col] = (_Float16)v;
        }
    }
}

// ---------------------------------------------------------------------------
// Host launcher
// ---------------------------------------------------------------------------
extern "C" void kernel_launch(void* const* d_in, const int* in_sizes, int n_in,
                              void* d_out, int out_size, void* d_ws, size_t ws_size,
                              hipStream_t stream) {
    const float* x     = (const float*)d_in[0];
    const float* mask  = (const float*)d_in[1];   // [B,L,1] -> per-token
    const float* gamma = (const float*)d_in[2];
    const float* beta  = (const float*)d_in[3];
    const float* Wq = (const float*)d_in[4];  const float* bq = (const float*)d_in[5];
    const float* Wk = (const float*)d_in[6];  const float* bk = (const float*)d_in[7];
    const float* Wv = (const float*)d_in[8];  const float* bv = (const float*)d_in[9];
    const float* W1 = (const float*)d_in[10]; const float* b1 = (const float*)d_in[11];
    const float* W2 = (const float*)d_in[12]; const float* b2 = (const float*)d_in[13];
    float* out = (float*)d_out;

    // workspace carve-out
    size_t off = 0;
    auto carve = [&](size_t bytes) {
        void* p = (char*)d_ws + off;
        off += (bytes + 255) & ~(size_t)255;
        return p;
    };
    const size_t fAct = (size_t)kTokens * kD * sizeof(float);
    const size_t hAct = (size_t)kTokens * kD * sizeof(_Float16);
    const size_t hW   = (size_t)kD * kD * sizeof(_Float16);

    float*    xnf   = (float*)carve(fAct);
    float*    attnF = (float*)carve(fAct);
    _Float16* xnh   = (_Float16*)carve(hAct);
    _Float16* xh    = (_Float16*)carve(hAct);
    _Float16* Qh    = (_Float16*)carve(hAct);
    _Float16* Khb   = (_Float16*)carve(hAct);
    _Float16* Vhb   = (_Float16*)carve(hAct);
    _Float16* attnH = (_Float16*)carve(hAct);
    _Float16* ffnH  = (_Float16*)carve(hAct);
    _Float16* WqH   = (_Float16*)carve(hW);
    _Float16* WkH   = (_Float16*)carve(hW);
    _Float16* WvH   = (_Float16*)carve(hW);
    _Float16* W1H   = (_Float16*)carve(hW);
    _Float16* W2H   = (_Float16*)carve(hW);
    float*    keym  = (float*)carve((size_t)kTokens * sizeof(float));
    float*    qm    = (float*)carve((size_t)kTokens * sizeof(float));

    // 1) layernorm + masks + f16 staging
    prep_kernel<<<kTokens / 4, 128, 0, stream>>>(x, gamma, beta, xnf, xnh, xh, keym, qm);

    // 2) weight conversion
    const int wn = kD * kD;
    cvt_kernel<<<(wn + 255) / 256, 256, 0, stream>>>(Wq, WqH, wn);
    cvt_kernel<<<(wn + 255) / 256, 256, 0, stream>>>(Wk, WkH, wn);
    cvt_kernel<<<(wn + 255) / 256, 256, 0, stream>>>(Wv, WvH, wn);
    cvt_kernel<<<(wn + 255) / 256, 256, 0, stream>>>(W1, W1H, wn);
    cvt_kernel<<<(wn + 255) / 256, 256, 0, stream>>>(W2, W2H, wn);

    // 3) projections (Q from x_norm; K,V from x — matches reference)
    const int gblocks = (kTokens / 64) * (kD / 64);
    gemm_kernel<<<gblocks, 128, 0, stream>>>(xnh, WqH, bq, nullptr, nullptr, Qh,  nullptr,
                                             kTokens, kD, kD, 0);
    gemm_kernel<<<gblocks, 128, 0, stream>>>(xh,  WkH, bk, nullptr, nullptr, Khb, nullptr,
                                             kTokens, kD, kD, 0);
    gemm_kernel<<<gblocks, 128, 0, stream>>>(xh,  WvH, bv, nullptr, nullptr, Vhb, nullptr,
                                             kTokens, kD, kD, 0);

    // 4) attention (+ x_norm residual)
    attn_kernel<<<32 * kHeads * (kSeq / 16), 32, 0, stream>>>(
        Qh, Khb, Vhb, keym, qm, xnf, attnF, attnH);

    // 5) FFN: relu(attn@W1^T+b1) @ W2^T + b2 + attn, then * mask
    gemm_kernel<<<gblocks, 128, 0, stream>>>(attnH, W1H, b1, nullptr, nullptr, ffnH, nullptr,
                                             kTokens, kD, kD, 1);
    gemm_kernel<<<gblocks, 128, 0, stream>>>(ffnH, W2H, b2, attnF, mask, nullptr, out,
                                             kTokens, kD, kD, 2);
    (void)in_sizes; (void)n_in; (void)out_size; (void)ws_size;
}
